// MultiHeadAttentionRoPE_82265803587779
// MI455X (gfx1250) — matrix operand order repaired and checked by reading, not compiled, not run on hardware
//
#include <hip/hip_runtime.h>

typedef __attribute__((ext_vector_type(16))) _Float16 v16h;
typedef __attribute__((ext_vector_type(8)))  float    v8f;
typedef __attribute__((ext_vector_type(4)))  int      v4i;

#define WMMA(a, b, c) __builtin_amdgcn_wmma_f32_16x16x32_f16(false, (a), false, (b), (short)0, (c), false, false)

namespace {
constexpr int Bc = 2, Tc = 2048, Dc = 1024, Hc = 16;
constexpr int BTc  = Bc * Tc;    // 4096 rows
constexpr int MTc  = BTc / 16;   // 256 row tiles over BT
constexpr int KTDc = Dc / 32;    // 32 K-tiles over D
constexpr int NTDc = Dc / 16;    // 64 N-tiles over D
constexpr int TT16c = Tc / 16;   // 128
constexpr int TT32c = Tc / 32;   // 64
}

// ---- gfx1250 async global->LDS copy (16B per call), with portable fallback ----
#if __has_builtin(__builtin_amdgcn_global_load_async_to_lds_b128)
#define HAVE_ASYNC_LDS 1
typedef __attribute__((address_space(1))) v4i gv4i;
typedef __attribute__((address_space(3))) v4i lv4i;
#endif

__device__ __forceinline__ void async_cp16(const void* g, void* l) {
#ifdef HAVE_ASYNC_LDS
  __builtin_amdgcn_global_load_async_to_lds_b128((gv4i*)(g), (lv4i*)(l), 0, 0);
#else
  *(v4i*)l = *(const v4i*)g;
#endif
}

__device__ __forceinline__ void async_wait0() {
#ifdef HAVE_ASYNC_LDS
#if __has_builtin(__builtin_amdgcn_s_wait_asynccnt)
  __builtin_amdgcn_s_wait_asynccnt(0);
#else
  asm volatile("s_wait_asynccnt 0" ::: "memory");
#endif
#endif
}

// fp32 row-major [BT x D] -> f16 A-fragment-contiguous tiles (16x32 per tile).
// A-frag mapping (16-bit A 16x32): lane = half*16 + m, elem e -> k = (e/8)*16 + half*8 + (e%8).
__global__ __launch_bounds__(256) void k_swizzleA(const float* __restrict__ src,
                                                  _Float16* __restrict__ dst) {
  int idx  = blockIdx.x * 256 + threadIdx.x;
  int e    = idx & 15;
  int lane = (idx >> 4) & 31;
  int tile = idx >> 9;
  int kt = tile % KTDc;
  int mt = tile / KTDc;
  int m = mt * 16 + (lane & 15);
  int k = kt * 32 + ((e >> 3) << 4) + ((lane >> 4) << 3) + (e & 7);
  dst[idx] = (_Float16)src[(size_t)m * Dc + k];
}

// fp32 row-major weight [D x D] (rows = K dim) -> f16 B-fragment-contiguous tiles (32x16).
// B-frag mapping (16-bit B 32x16): lane = half*16 + n, elem e -> k = (e/8)*16 + half*8 + (e%8).
__global__ __launch_bounds__(256) void k_swizzleB(const float* __restrict__ src,
                                                  _Float16* __restrict__ dst) {
  int idx  = blockIdx.x * 256 + threadIdx.x;
  int e    = idx & 15;
  int lane = (idx >> 4) & 31;
  int tile = idx >> 9;
  int nt = tile % NTDc;
  int kt = tile / NTDc;
  int n = nt * 16 + (lane & 15);
  int k = kt * 32 + ((e >> 3) << 4) + ((lane >> 4) << 3) + (e & 7);
  dst[idx] = (_Float16)src[(size_t)k * Dc + n];
}

// Q/K/V projection + RoPE. 4 waves/WG: each wave a 16-row tile; the 12KB B panel
// (wq/wk/wv x 4 head n-tiles) per K-step is async-staged in LDS, double-buffered.
__global__ __launch_bounds__(128) void k_qkv_rope(const _Float16* __restrict__ xs,
                                                  const _Float16* __restrict__ wqs,
                                                  const _Float16* __restrict__ wks,
                                                  const _Float16* __restrict__ wvs,
                                                  _Float16* __restrict__ qs,
                                                  _Float16* __restrict__ ks,
                                                  _Float16* __restrict__ vs) {
  __shared__ _Float16 pan[2][3 * 2048];        // [buf][matrix*2048 + nt*512 + lane*16 + e]
  const int tid  = threadIdx.x;
  const int lane = tid & 31;
  const int wv_  = tid >> 5;                   // wave id 0..3
  const int mt = blockIdx.x * 4 + wv_;
  const int h  = blockIdx.y;
  const _Float16* Wsrc[3] = {wqs, wks, wvs};

  auto issue = [&](int kt, int buf) {
    const size_t base = ((size_t)kt * NTDc + h * 4) * 512;  // 2048 contiguous halfs per matrix
#pragma unroll
    for (int m = 0; m < 3; ++m) {
      const char* g = (const char*)(Wsrc[m] + base) + tid * 32;
      char* l = (char*)&pan[buf][m * 2048] + tid * 32;
      async_cp16(g, l);
      async_cp16(g + 16, l + 16);
    }
  };

  v8f cq[4] = {}; v8f ck[4] = {}; v8f cv[4] = {};
  const v16h* A = (const v16h*)xs + (size_t)mt * KTDc * 32;

  issue(0, 0);
  for (int kt = 0; kt < KTDc; ++kt) {
    async_wait0();
    __syncthreads();
    if (kt + 1 < KTDc) issue(kt + 1, (kt + 1) & 1);
    const int buf = kt & 1;
    v16h a = A[kt * 32 + lane];
    const v16h* P = (const v16h*)&pan[buf][0];   // 128 v16h per matrix
#pragma unroll
    for (int nt = 0; nt < 4; ++nt) {
      cq[nt] = WMMA(a, P[0 * 128 + nt * 32 + lane], cq[nt]);
      ck[nt] = WMMA(a, P[1 * 128 + nt * 32 + lane], ck[nt]);
      cv[nt] = WMMA(a, P[2 * 128 + nt * 32 + lane], cv[nt]);
    }
  }

  const int half = lane >> 4;
  const int n16  = lane & 15;
  const int rowbase = mt * 16;
  const int b     = rowbase / Tc;
  const int tbase = rowbase % Tc;
  const size_t bh = (size_t)b * Hc + h;
#pragma unroll
  for (int nt = 0; nt < 4; ++nt) {
#pragma unroll
    for (int r = 0; r < 8; ++r) {
      const int hd   = nt * 16 + n16;
      const int tloc = tbase + r + 8 * half;
      // RoPE: inv_freq = 10000^(-(hd%32)/32); partner (hd +- 32) lives in tile nt^2.
      const float inv = __expf(-0.28782313662425572f * (float)(hd & 31));
      float sn, cs;
      __sincosf((float)tloc * inv, &sn, &cs);
      const float vq = cq[nt][r], pq = cq[nt ^ 2][r];
      const float vk = ck[nt][r], pk = ck[nt ^ 2][r];
      const float qv = vq * cs + ((hd < 32) ? -pq : pq) * sn;
      const float kv = vk * cs + ((hd < 32) ? -pk : pk) * sn;
      { // Q: A-fragment layout over (t, hd): [b][h][T/16][2][32][16]
        int eo = (((hd & 31) >> 4) << 3) + (hd & 7);
        int ln = (((hd & 15) >> 3) << 4) + (tloc & 15);
        size_t off = (((bh * TT16c + (tloc >> 4)) * 2 + (hd >> 5)) * 32 + ln) * 16 + eo;
        qs[off] = (_Float16)qv;
      }
      { // K^T: B-fragment layout, Kdim = hd, N = key t: [b][h][2][T/16][32][16]
        int ln = (((hd & 15) >> 3) << 4) + (tloc & 15);
        int eo = (((hd & 31) >> 4) << 3) + (hd & 7);
        size_t off = (((bh * 2 + (hd >> 5)) * TT16c + (tloc >> 4)) * 32 + ln) * 16 + eo;
        ks[off] = (_Float16)kv;
      }
      { // V: B-fragment layout, Kdim = key t, N = hd: [b][h][T/32][4][32][16]
        int ln = (((tloc & 15) >> 3) << 4) + (hd & 15);
        int eo = (((tloc & 31) >> 4) << 3) + (tloc & 7);
        size_t off = (((bh * TT32c + (tloc >> 5)) * 4 + (hd >> 4)) * 32 + ln) * 16 + eo;
        vs[off] = (_Float16)cv[nt][r];
      }
    }
  }
}

// Flash attention. 4 waves/WG handle 4 consecutive query tiles sharing one K/V
// stream: 8KB K/V panel per 32-key block async-staged in LDS (double-buffered).
__global__ __launch_bounds__(128) void k_attn(const _Float16* __restrict__ qs,
                                              const _Float16* __restrict__ ks,
                                              const _Float16* __restrict__ vs,
                                              _Float16* __restrict__ ctxs) {
  __shared__ _Float16 kpan[2][2 * 1024];   // [buf][khd*1024 + j*512 + lane*16 + e]
  __shared__ _Float16 vpan[2][2048];       // [buf][jt*512 + lane*16 + e]
  __shared__ _Float16 Pl[4][16 * 32];      // per-wave P tile
  const int tid  = threadIdx.x;
  const int lane = tid & 31;
  const int wv_  = tid >> 5;
  const int qg = blockIdx.x;               // query-tile group of 4
  const int h  = blockIdx.y;
  const int b  = blockIdx.z;
  const int qt = qg * 4 + wv_;
  const int half = lane >> 4;
  const int n16  = lane & 15;
  const size_t bh = (size_t)b * Hc + h;

  const v16h* Q  = (const v16h*)qs + ((bh * TT16c + qt) * 2) * 32;
  const v16h a0 = Q[lane];
  const v16h a1 = Q[32 + lane];
  const char* Kb = (const char*)((const v16h*)ks + (bh * 2) * TT16c * 32);   // tile = 1KB
  const char* Vb = (const char*)((const v16h*)vs + (bh * TT32c) * 4 * 32);

  auto issue = [&](int kb, int buf) {
#pragma unroll
    for (int khd = 0; khd < 2; ++khd) {    // 2KB contiguous per khd: tiles (2kb, 2kb+1)
      const char* g = Kb + ((size_t)(khd * TT16c + 2 * kb)) * 1024 + tid * 16;
      async_cp16(g, (char*)&kpan[buf][khd * 1024] + tid * 16);
    }
    const char* g = Vb + (size_t)(kb * 4) * 1024 + tid * 32;   // 4KB contiguous
    char* l = (char*)&vpan[buf][0] + tid * 32;
    async_cp16(g, l);
    async_cp16(g + 16, l + 16);
  };

  v8f o[4] = {};
  float runm[8], runl[8];
#pragma unroll
  for (int r = 0; r < 8; ++r) { runm[r] = -1e30f; runl[r] = 0.f; }

  const int qr_base = qt * 16 + 8 * half;
  const int nkb_w   = (qt * 16 + 47) >> 5;       // this wave's causal key-block count
  const int nkb_max = (qg * 64 + 95) >> 5;       // workgroup-uniform trip count

  issue(0, 0);
  for (int kb = 0; kb < nkb_max; ++kb) {
    async_wait0();
    __syncthreads();
    if (kb + 1 < nkb_max) issue(kb + 1, (kb + 1) & 1);
    if (kb < nkb_w) {                             // wave-uniform predicate
      const int buf = kb & 1;
      v8f s0 = {}, s1 = {};
      s0 = WMMA(a0, *(const v16h*)&kpan[buf][0 * 1024 + 0 * 512 + lane * 16], s0);
      s0 = WMMA(a1, *(const v16h*)&kpan[buf][1 * 1024 + 0 * 512 + lane * 16], s0);
      s1 = WMMA(a0, *(const v16h*)&kpan[buf][0 * 1024 + 1 * 512 + lane * 16], s1);
      s1 = WMMA(a1, *(const v16h*)&kpan[buf][1 * 1024 + 1 * 512 + lane * 16], s1);
      const int key0 = kb * 32 + n16;
      const int key1 = key0 + 16;
#pragma unroll
      for (int r = 0; r < 8; ++r) {
        const int qr = qr_base + r;
        float e0 = (key0 > qr) ? -1e30f : s0[r] * 0.125f;   // 1/sqrt(64)
        float e1 = (key1 > qr) ? -1e30f : s1[r] * 0.125f;
        float mx = fmaxf(e0, e1);
        mx = fmaxf(mx, __shfl_xor(mx, 1));
        mx = fmaxf(mx, __shfl_xor(mx, 2));
        mx = fmaxf(mx, __shfl_xor(mx, 4));
        mx = fmaxf(mx, __shfl_xor(mx, 8));
        const float nm = fmaxf(runm[r], mx);
        const float alpha = __expf(runm[r] - nm);
        runm[r] = nm;
        const float p0 = __expf(e0 - nm);
        const float p1 = __expf(e1 - nm);
        float rs = p0 + p1;
        rs += __shfl_xor(rs, 1);
        rs += __shfl_xor(rs, 2);
        rs += __shfl_xor(rs, 4);
        rs += __shfl_xor(rs, 8);
        runl[r] = runl[r] * alpha + rs;
#pragma unroll
        for (int j = 0; j < 4; ++j) o[j][r] *= alpha;
        const int prow = r + 8 * half;
        Pl[wv_][prow * 32 + n16]      = (_Float16)p0;
        Pl[wv_][prow * 32 + 16 + n16] = (_Float16)p1;
      }
      asm volatile("s_wait_dscnt 0" ::: "memory");   // cross-lane P visibility (per wave)
      v16h ap;
#pragma unroll
      for (int e = 0; e < 16; ++e) {
        const int kk = ((e >> 3) << 4) + (half << 3) + (e & 7);
        ap[e] = Pl[wv_][n16 * 32 + kk];
      }
#pragma unroll
      for (int j = 0; j < 4; ++j)
        o[j] = WMMA(ap, *(const v16h*)&vpan[buf][j * 512 + lane * 16], o[j]);
    }
  }

  // ctx -> A-fragment layout over (BT rows, D cols) for the output projection
  const int mt = b * TT16c + qt;
#pragma unroll
  for (int j = 0; j < 4; ++j) {
#pragma unroll
    for (int r = 0; r < 8; ++r) {
      const float v = o[j][r] / runl[r];
      const int d  = h * 64 + j * 16 + n16;
      const int eo = (((d & 31) >> 4) << 3) + (d & 7);
      const int ln = (((d & 15) >> 3) << 4) + (r + 8 * half);
      const size_t off = (((size_t)mt * KTDc + (d >> 5)) * 32 + ln) * 16 + eo;
      ctxs[off] = (_Float16)v;
    }
  }
}

// Output projection: 4 waves/WG share the async-staged w_o panel.
__global__ __launch_bounds__(128) void k_oproj(const _Float16* __restrict__ ctxs,
                                               const _Float16* __restrict__ wos,
                                               float* __restrict__ out) {
  __shared__ _Float16 pan[2][2048];
  const int tid  = threadIdx.x;
  const int lane = tid & 31;
  const int wv_  = tid >> 5;
  const int mt = blockIdx.x * 4 + wv_;
  const int ng = blockIdx.y;               // group of 4 N-tiles (64 output cols)

  auto issue = [&](int kt, int buf) {
    const char* g = (const char*)(wos + ((size_t)kt * NTDc + ng * 4) * 512) + tid * 32;
    char* l = (char*)&pan[buf][0] + tid * 32;
    async_cp16(g, l);
    async_cp16(g + 16, l + 16);
  };

  v8f c[4] = {};
  const v16h* A = (const v16h*)ctxs + (size_t)mt * KTDc * 32;

  issue(0, 0);
  for (int kt = 0; kt < KTDc; ++kt) {
    async_wait0();
    __syncthreads();
    if (kt + 1 < KTDc) issue(kt + 1, (kt + 1) & 1);
    const int buf = kt & 1;
    v16h a = A[kt * 32 + lane];
#pragma unroll
    for (int j = 0; j < 4; ++j)
      c[j] = WMMA(a, *(const v16h*)&pan[buf][j * 512 + lane * 16], c[j]);
  }
  const int half = lane >> 4, n16 = lane & 15;
#pragma unroll
  for (int j = 0; j < 4; ++j)
#pragma unroll
    for (int r = 0; r < 8; ++r)
      out[(size_t)(mt * 16 + r + 8 * half) * Dc + ng * 64 + j * 16 + n16] = c[j][r];
}

extern "C" void kernel_launch(void* const* d_in, const int* in_sizes, int n_in,
                              void* d_out, int out_size, void* d_ws, size_t ws_size,
                              hipStream_t stream) {
  (void)in_sizes; (void)n_in; (void)out_size; (void)ws_size;
  const float* x  = (const float*)d_in[0];
  const float* wq = (const float*)d_in[1];
  const float* wk = (const float*)d_in[2];
  const float* wv = (const float*)d_in[3];
  const float* wo = (const float*)d_in[4];
  float* out = (float*)d_out;

  // Workspace carve-up (f16 halves): ~48 MB total.
  _Float16* p = (_Float16*)d_ws;
  _Float16* xs   = p; p += (size_t)BTc * Dc;
  _Float16* wqs  = p; p += (size_t)Dc * Dc;
  _Float16* wks  = p; p += (size_t)Dc * Dc;
  _Float16* wvs  = p; p += (size_t)Dc * Dc;
  _Float16* wos  = p; p += (size_t)Dc * Dc;
  _Float16* qsb  = p; p += (size_t)BTc * Dc;
  _Float16* ksb  = p; p += (size_t)BTc * Dc;
  _Float16* vsb  = p; p += (size_t)BTc * Dc;
  _Float16* ctxs = p; p += (size_t)BTc * Dc;

  k_swizzleA<<<(BTc * Dc) / 256, 256, 0, stream>>>(x, xs);
  k_swizzleB<<<(Dc * Dc) / 256, 256, 0, stream>>>(wq, wqs);
  k_swizzleB<<<(Dc * Dc) / 256, 256, 0, stream>>>(wk, wks);
  k_swizzleB<<<(Dc * Dc) / 256, 256, 0, stream>>>(wv, wvs);
  k_swizzleB<<<(Dc * Dc) / 256, 256, 0, stream>>>(wo, wos);
  k_qkv_rope<<<dim3(MTc / 4, Hc), 128, 0, stream>>>(xs, wqs, wks, wvs, qsb, ksb, vsb);
  k_attn<<<dim3(TT16c / 4, Hc, Bc), 128, 0, stream>>>(qsb, ksb, vsb, ctxs);
  k_oproj<<<dim3(MTc / 4, Dc / 64), 128, 0, stream>>>(ctxs, wos, out);
}
